// LSTM_44092134261579
// MI455X (gfx1250) — matrix operand order repaired and checked by reading, not compiled, not run on hardware
//
#include <hip/hip_runtime.h>

typedef __attribute__((ext_vector_type(16))) _Float16 v16h;
typedef __attribute__((ext_vector_type(8)))  float    v8f;

#define DEVINL __device__ __forceinline__

// CDNA5 async global->LDS staging (GLOBAL_LOAD_ASYNC_TO_LDS_B32, ASYNCcnt).
#define USE_ASYNC_LDS 1

namespace {

constexpr int BATCH = 4096;
constexpr int T     = 512;
constexpr int H     = 32;
constexpr int NOUT  = 12;
constexpr int TS    = 40;   // padded LDS row stride (halves): bank-conflict-free b128

DEVINL float tanh_f(float x) {
#if __has_builtin(__builtin_amdgcn_tanhf)
  return __builtin_amdgcn_tanhf(x);           // native v_tanh_f32
#else
  float e = __builtin_amdgcn_exp2f(2.8853900817779268f * x);
  return 1.0f - 2.0f * __builtin_amdgcn_rcpf(e + 1.0f);
#endif
}

// sigmoid(x + b) = 0.5 + 0.5*tanh(0.5*x + 0.5*b); pass hb = 0.5*b. One TRANS op.
DEVINL float sigmoid_b(float x, float hb) {
  return __builtin_fmaf(tanh_f(__builtin_fmaf(x, 0.5f, hb)), 0.5f, 0.5f);
}

// B-fragment (K=32 x N=16, f16): lane n=lane&15 -> column (gate row of W),
// kb=(lane>>4)*16 -> 16 consecutive K. W [128,32] row-major: 16 contiguous floats.
DEVINL v16h load_bfrag(const float* __restrict__ p) {
  v16h r;
#pragma unroll
  for (int j = 0; j < 16; ++j) r[j] = (_Float16)p[j];
  return r;
}

// A-fragment (M=16 x K=32, f16) from padded LDS tile [16][TS].
DEVINL v16h load_afrag(const _Float16* tile, int m, int s) {
  v16h a;
#pragma unroll
  for (int j = 0; j < 8; ++j) {
    a[j]     = tile[m * TS + s + j];
    a[j + 8] = tile[m * TS + s + 8 + j];
  }
  return a;
}

DEVINL v8f wmma_f16(v16h a, v16h b, v8f c) {
  return __builtin_amdgcn_wmma_f32_16x16x32_f16(false, a, false, b, (short)0, c,
                                                false, false);
}

DEVINL void stage_x(const float* gaddr, float* laddr) {
#if USE_ASYNC_LDS
  asm volatile("global_load_async_to_lds_b32 %0, %1, off"
               :: "v"((unsigned)(uintptr_t)laddr),
                  "v"((unsigned long long)(uintptr_t)gaddr)
               : "memory");
#else
  *laddr = *gaddr;   // plain vmem load + ds store fallback
#endif
}

DEVINL void wait_x(int pending) {
#if USE_ASYNC_LDS
  if (pending) asm volatile("s_wait_asynccnt 0x1" ::: "memory");
  else         asm volatile("s_wait_asynccnt 0x0" ::: "memory");
#else
  (void)pending;
#endif
}

} // namespace

// Block = 64 threads = 2 waves, one 16-row batch tile. Wave wv owns hidden
// units 16*wv..16*wv+15; gate q (i,f,g,o) of those units lives at column
// 16*(2q+wv)+n of the [*,128] gate matrix. Layer 1 is software-pipelined one
// step behind layer 0 so each iteration needs a single workgroup barrier.
__global__ __launch_bounds__(64) void lstm2_wmma_kernel(
    const float* __restrict__ x, const int* __restrict__ lengths,
    const float* __restrict__ Wih0, const float* __restrict__ Whh0,
    const float* __restrict__ bih0, const float* __restrict__ bhh0,
    const float* __restrict__ Wih1, const float* __restrict__ Whh1,
    const float* __restrict__ bih1, const float* __restrict__ bhh1,
    const float* __restrict__ Wp, const float* __restrict__ bp,
    const float* __restrict__ Wv, const float* __restrict__ bv,
    float* __restrict__ out)
{
  __shared__ _Float16 h0A[2][16 * TS];         // layer-0 hidden tile, ping-pong
  __shared__ _Float16 h1A[2][16 * TS];         // layer-1 hidden tile, ping-pong
  __shared__ float    xstg[2][2][2][16];       // [wave][buf][t parity][row]

  const int lane = threadIdx.x & 31;
  const int wv   = threadIdx.x >> 5;
  const int n    = lane & 15;
  const int hi   = lane >> 4;
  const int m    = n;             // A-fragment row
  const int s    = 16 * hi;       // A-fragment K base
  const int kb   = 16 * hi;       // B-fragment K base
  const int rowb = blockIdx.x * 16;

  // ---- weights: 12 B-fragments resident in VGPRs for the whole recurrence ----
  v16h Bh0[4], Bi1[4], Bh1[4];
  float b0s[4], b1s[4], wi0g[4];
#pragma unroll
  for (int q = 0; q < 4; ++q) {
    const int col = 16 * (2 * q + wv) + n;
    Bh0[q] = load_bfrag(Whh0 + col * H + kb);
    Bi1[q] = load_bfrag(Wih1 + col * H + kb);
    Bh1[q] = load_bfrag(Whh1 + col * H + kb);
    const float b0 = bih0[col] + bhh0[col];
    const float b1 = bih1[col] + bhh1[col];
    b0s[q] = (q == 2) ? b0 : 0.5f * b0;        // g-gate keeps full bias
    b1s[q] = (q == 2) ? b1 : 0.5f * b1;
    wi0g[q] = Wih0[col];                       // IN == 1
  }

  int lenr[8];
#pragma unroll
  for (int r = 0; r < 8; ++r) lenr[r] = lengths[rowb + 8 * hi + r];

  // tile-wide max length -> early exit (results past it are frozen anyway)
  int lm = lenr[0];
#pragma unroll
  for (int r = 1; r < 8; ++r) lm = max(lm, lenr[r]);
  lm = max(lm, __shfl_xor(lm, 16, 32));
  const int tmax = __builtin_amdgcn_readfirstlane(lm);

  // zero only h1A[0] (first a1h read); everything else is written before read
  {
    _Float16* p = &h1A[0][0];
#pragma unroll
    for (int j = 0; j < 10; ++j) p[threadIdx.x * 10 + j] = (_Float16)0.0f;
  }

  // x staging: lane covers (row = n, t-parity = hi); one async b32 per 2 steps
  const float* gx = x + (size_t)(rowb + n) * T + hi;
  float* lx[2] = { &xstg[wv][0][hi][n], &xstg[wv][1][hi][n] };
  stage_x(gx + 0, lx[0]);                      // pair (t=0, t=1) -> buffer 0

  float c0[8] = {0}, h0[8] = {0}, c1[8] = {0}, h1[8] = {0};
  bool  lv[8];                                 // layer-0 liveness at its last t
  v8f zacc;
#pragma unroll
  for (int r = 0; r < 8; ++r) zacc[r] = 0.0f;

  // ---- per-layer step bodies ----
  auto layer0_step = [&](int t, const v16h& aA, const float* xr) {
    v8f g0[4];
#pragma unroll
    for (int q = 0; q < 4; ++q) {
#pragma unroll
      for (int r = 0; r < 8; ++r) g0[q][r] = wi0g[q] * xr[r];  // bias folded later
      g0[q] = wmma_f16(aA, Bh0[q], g0[q]);
    }
#pragma unroll
    for (int r = 0; r < 8; ++r) {
      float ig = sigmoid_b(g0[0][r], b0s[0]);
      float fg = sigmoid_b(g0[1][r], b0s[1]);
      float gg = tanh_f(g0[2][r] + b0s[2]);
      float og = sigmoid_b(g0[3][r], b0s[3]);
      float cn = __builtin_fmaf(fg, c0[r], ig * gg);
      c0[r] = cn;                              // dead rows never committed via h
      float hn = og * tanh_f(cn);
      bool nl  = (t < lenr[r]);
      h0[r] = nl ? hn : h0[r];
      lv[r] = nl;
      h0A[t & 1][(8 * hi + r) * TS + 16 * wv + n] = (_Float16)h0[r];
    }
  };

  auto layer1_step = [&](int t, const v16h& aX, const v16h& aH, bool store) {
    v8f g1[4];
#pragma unroll
    for (int q = 0; q < 4; ++q) {
      g1[q] = wmma_f16(aX, Bi1[q], zacc);
      g1[q] = wmma_f16(aH, Bh1[q], g1[q]);
    }
#pragma unroll
    for (int r = 0; r < 8; ++r) {
      float ig = sigmoid_b(g1[0][r], b1s[0]);
      float fg = sigmoid_b(g1[1][r], b1s[1]);
      float gg = tanh_f(g1[2][r] + b1s[2]);
      float og = sigmoid_b(g1[3][r], b1s[3]);
      float cn = __builtin_fmaf(fg, c1[r], ig * gg);
      c1[r] = cn;
      float hn = og * tanh_f(cn);
      h1[r] = lv[r] ? hn : h1[r];              // lv holds (t < lenr) from layer 0
      if (store)
        h1A[(t + 1) & 1][(8 * hi + r) * TS + 16 * wv + n] = (_Float16)h1[r];
    }
  };

  __syncthreads();                             // h1A[0] zero-init visible

  // ---- pipeline prologue: iteration 0 runs layer0(t=0) only ----
  {
    if (2 < tmax) { stage_x(gx + 2, lx[1]); wait_x(1); }
    else          { wait_x(0); }
    const float* xs = &xstg[wv][0][0][8 * hi];
    float xr[8];
#pragma unroll
    for (int r = 0; r < 8; ++r) xr[r] = xs[r];
    v16h az;
#pragma unroll
    for (int j = 0; j < 16; ++j) az[j] = (_Float16)0.0f;
    layer0_step(0, az, xr);
    __syncthreads();
  }

  // ---- main pipeline: iteration i runs layer1(i-1) + layer0(i) ----
#pragma unroll 2
  for (int i = 1; i < tmax; ++i) {
    const int pb = (i >> 1) & 1;
    if ((i & 1) == 0) {
      if (i + 2 < tmax) { stage_x(gx + i + 2, lx[pb ^ 1]); wait_x(1); }
      else              { wait_x(0); }
    }
    const float* xs = &xstg[wv][pb][i & 1][8 * hi];
    float xr[8];
#pragma unroll
    for (int r = 0; r < 8; ++r) xr[r] = xs[r];

    // h0 after step i-1: A operand for layer0(i) AND input operand for layer1(i-1)
    v16h ax  = load_afrag(h0A[(i - 1) & 1], m, s);
    v16h a1h = load_afrag(h1A[(i - 1) & 1], m, s);

    layer1_step(i - 1, ax, a1h, true);
    layer0_step(i, ax, xr);
    __syncthreads();
  }

  // ---- pipeline epilogue: layer1(tmax-1) ----
  {
    v16h ax  = load_afrag(h0A[(tmax - 1) & 1], m, s);
    v16h a1h = load_afrag(h1A[(tmax - 1) & 1], m, s);
    layer1_step(tmax - 1, ax, a1h, false);
  }
  __syncthreads();                             // epilogue reads done before reuse

  // -------- heads: hT1 @ [Wp | Wv]^T (+ bias) in a single WMMA --------
#pragma unroll
  for (int r = 0; r < 8; ++r)
    h1A[0][(8 * hi + r) * TS + 16 * wv + n] = (_Float16)h1[r];
  __syncthreads();

  v16h aH = load_afrag(h1A[0], m, s);
  v16h Bhd;
#pragma unroll
  for (int j = 0; j < 16; ++j) {
    float w = 0.0f;
    if (n < NOUT)        w = Wp[n * H + kb + j];
    else if (n == NOUT)  w = Wv[kb + j];
    Bhd[j] = (_Float16)w;
  }
  const float hb = (n < NOUT) ? bp[n] : ((n == NOUT) ? bv[0] : 0.0f);
  v8f d;
#pragma unroll
  for (int r = 0; r < 8; ++r) d[r] = hb;
  d = wmma_f16(aH, Bhd, d);

  if (wv == 0) {
#pragma unroll
    for (int r = 0; r < 8; ++r) {
      const int b = rowb + 8 * hi + r;
      if (n < NOUT)       out[(size_t)b * NOUT + n] = d[r];
      else if (n == NOUT) out[(size_t)BATCH * NOUT + b] = d[r];
    }
  }
}

extern "C" void kernel_launch(void* const* d_in, const int* in_sizes, int n_in,
                              void* d_out, int out_size, void* d_ws, size_t ws_size,
                              hipStream_t stream) {
  (void)in_sizes; (void)n_in; (void)d_ws; (void)ws_size; (void)out_size;
  const float* xp   = (const float*)d_in[0];
  const int*   lenp = (const int*)  d_in[1];
  const float* Wih0 = (const float*)d_in[2];
  const float* Whh0 = (const float*)d_in[3];
  const float* bih0 = (const float*)d_in[4];
  const float* bhh0 = (const float*)d_in[5];
  const float* Wih1 = (const float*)d_in[6];
  const float* Whh1 = (const float*)d_in[7];
  const float* bih1 = (const float*)d_in[8];
  const float* bhh1 = (const float*)d_in[9];
  const float* Wp   = (const float*)d_in[10];
  const float* bp   = (const float*)d_in[11];
  const float* Wv   = (const float*)d_in[12];
  const float* bv   = (const float*)d_in[13];
  float* out = (float*)d_out;

  dim3 grid(BATCH / 16);   // 256 blocks
  dim3 block(64);          // 2 waves per tile
  hipLaunchKernelGGL(lstm2_wmma_kernel, grid, block, 0, stream,
                     xp, lenp, Wih0, Whh0, bih0, bhh0,
                     Wih1, Whh1, bih1, bhh1, Wp, bp, Wv, bv, out);
}
